// Attention_concat_45002667327581
// MI455X (gfx1250) — matrix operand order repaired and checked by reading, not compiled, not run on hardware
//
#include <hip/hip_runtime.h>

typedef __attribute__((ext_vector_type(16))) _Float16 v16h;
typedef __attribute__((ext_vector_type(8)))  _Float16 v8h;
typedef __attribute__((ext_vector_type(8)))  float    v8f;
typedef __attribute__((ext_vector_type(4)))  float    v4f;

#define NPIX 4096   // H*W
#define CCH  64
#define NB   4
#define NCH  (NPIX / 32)   // 128 K-chunks

// ---------------------------------------------------------------------------
// Kernel A: fold wcq/wck into the 1x1 conv weights; convert Wg to f16.
//   uq[c] = sum_o wcq[o]*Wq[o,c]   (a_n = uq . x_n + cq)
// ---------------------------------------------------------------------------
__global__ __launch_bounds__(64) void attn_setup(
    const float* __restrict__ Wq, const float* __restrict__ bq,
    const float* __restrict__ Wk, const float* __restrict__ bk,
    const float* __restrict__ wcq, const float* __restrict__ wck,
    const float* __restrict__ Wg,
    float* __restrict__ uq, float* __restrict__ uk, float* __restrict__ cqk,
    _Float16* __restrict__ wg16)
{
    const int t = threadIdx.x;   // 0..63 = channel
    float sq = 0.f, sk = 0.f;
    #pragma unroll
    for (int o = 0; o < 16; ++o) {
        sq = fmaf(wcq[o], Wq[o * CCH + t], sq);
        sk = fmaf(wck[o], Wk[o * CCH + t], sk);
    }
    uq[t] = sq;
    uk[t] = sk;
    if (t == 0) {
        float c1 = 0.f, c2 = 0.f;
        #pragma unroll
        for (int o = 0; o < 16; ++o) { c1 = fmaf(wcq[o], bq[o], c1); c2 = fmaf(wck[o], bk[o], c2); }
        cqk[0] = c1; cqk[1] = c2;
    }
    #pragma unroll 4
    for (int i = 0; i < CCH; ++i)
        wg16[t * CCH + i] = (_Float16)Wg[t * CCH + i];
}

// ---------------------------------------------------------------------------
// Kernel B: per pixel n: interleaved (a, exp(a)) and (d, exp(d)) pairs, and
// v = Wv x + bv stored f16 with layout v[b][c][n] (contiguous K per row).
// ---------------------------------------------------------------------------
__global__ __launch_bounds__(256) void attn_proj(
    const float* __restrict__ x, const float* __restrict__ Wv,
    const float* __restrict__ bv,
    const float* __restrict__ uq, const float* __restrict__ uk,
    const float* __restrict__ cqk,
    float2* __restrict__ aeO, float2* __restrict__ deO,
    _Float16* __restrict__ vO)
{
    const int b = blockIdx.y;
    const int n = blockIdx.x * 256 + threadIdx.x;
    const float* xb = x + (size_t)b * CCH * NPIX + n;

    float xr[CCH];
    float av = cqk[0], dv = cqk[1];
    #pragma unroll
    for (int c = 0; c < CCH; ++c) {
        float xv = xb[(size_t)c << 12];
        xr[c] = xv;
        av = fmaf(uq[c], xv, av);
        dv = fmaf(uk[c], xv, dv);
    }
    const int idx = b * NPIX + n;
    float2 ae; ae.x = av; ae.y = __expf(av);
    float2 de; de.x = dv; de.y = __expf(dv);
    aeO[idx] = ae;
    deO[idx] = de;

    for (int co = 0; co < CCH; ++co) {
        float acc = bv[co];
        #pragma unroll
        for (int c = 0; c < CCH; ++c)
            acc = fmaf(Wv[co * CCH + c], xr[c], acc);
        vO[((size_t)(b * CCH + co) << 12) + n] = (_Float16)acc;
    }
}

// ---------------------------------------------------------------------------
// Kernel C: fused  out = Wg @ ( (v @ E) * diag(1/(1.5*S)) ) + bg
// Grid: (N/128, B), block 256 (8 waves). Wave w owns columns j0+16w..+15,
// all 64 channels (4 row-blocks of 16x16).
// E[i,j] = elu(a_i+d_j) built in registers per K-chunk of 32 (no N^2 traffic).
// v chunks double-buffered in LDS; next chunk's global load is issued before
// the current chunk's WMMAs so the load latency overlaps compute.
// ---------------------------------------------------------------------------
__global__ __launch_bounds__(256) void attn_main(
    const float2* __restrict__ ae, const float2* __restrict__ de,
    const _Float16* __restrict__ v16, const _Float16* __restrict__ wg16,
    const float* __restrict__ bg, float* __restrict__ out)
{
    __shared__ _Float16 vtile[2][CCH * 32];   // 2 x 4 KB v chunk [c][k]
    __shared__ _Float16 wgS[CCH * CCH];       // 8 KB  Wg f16 row-major
    __shared__ _Float16 TtS[8 * 16 * CCH];    // 16 KB per-wave T^T [j][c]
    __shared__ float    bgS[CCH];

    const int tid  = threadIdx.x;
    const int wave = tid >> 5, lane = tid & 31;
    const int l15  = lane & 15, lhi = lane >> 4;   // column-in-tile, K-half
    const int b    = blockIdx.y;
    const int j    = blockIdx.x * 128 + wave * 16 + l15;

    // preload Wg (each thread 32B) and bg
    {
        const v8h* src = (const v8h*)wg16;
        v8h* dst = (v8h*)wgS;
        dst[tid * 2]     = src[tid * 2];
        dst[tid * 2 + 1] = src[tid * 2 + 1];
    }
    if (tid < CCH) bgS[tid] = bg[tid];

    const float2 dde = de[b * NPIX + j];
    const float dj = dde.x, edj = dde.y;
    // (a, exp(a)) pairs viewed as indivisible 16B vectors: a t index m4 covers
    // pairs (2*m4, 2*m4+1):  [a0, ea0, a1, ea1]
    const v4f* aev = (const v4f*)(ae + (size_t)b * NPIX);

    // staging identity: thread = (row cstage, 16B chunk chs)
    const int cstage = tid >> 2, chs = tid & 3;
    const v8h* gsrc = (const v8h*)(v16 + (((size_t)(b * CCH + cstage)) << 12));

    v8f acc[4] = {};
    float ssa[4] = {0.f, 0.f, 0.f, 0.f};

    // prologue: stage chunk 0
    v8h vreg = gsrc[chs];
    ((v8h*)vtile[0])[cstage * 4 + chs] = vreg;
    __syncthreads();

    for (int it = 0; it < NCH; ++it) {
        const int cur = it & 1, nxt = cur ^ 1;

        // issue next chunk's global load early (overlaps with compute below)
        if (it + 1 < NCH) vreg = gsrc[(it + 1) * 4 + chs];
        if (it + 2 < NCH) __builtin_prefetch((const void*)(gsrc + (it + 2) * 4 + chs), 0, 1);

        // B fragment: element m <-> K = lhi*16 + m, column = l15 (contiguous K order, same as A)
        // 8 contiguous b128 vector loads; branchless elu via v_cndmask.
        v16h bf;
        {
            const v4f* aep = aev + (it * 32 + lhi * 16) / 2;
            #pragma unroll
            for (int m4 = 0; m4 < 8; ++m4) {
                const v4f q = aep[m4];      // a[2m], ea[2m], a[2m+1], ea[2m+1]
                const float s0 = q[0] + dj;
                const float t0 = fmaf(q[1], edj, -1.f);
                const float e0 = (s0 > 0.f) ? s0 : t0;
                const float s1 = q[2] + dj;
                const float t1 = fmaf(q[3], edj, -1.f);
                const float e1 = (s1 > 0.f) ? s1 : t1;
                ssa[(m4 & 1) * 2 + 0] += e0;   // 4 independent partial sums
                ssa[(m4 & 1) * 2 + 1] += e1;
                bf[2 * m4]     = (_Float16)e0;
                bf[2 * m4 + 1] = (_Float16)e1;
            }
        }

        // A fragments (v rows) + 4 WMMAs
        #pragma unroll
        for (int rb = 0; rb < 4; ++rb) {
            const int c = rb * 16 + l15;
            const _Float16* base = vtile[cur] + c * 32 + lhi * 16; // halves
            v8h lo = *(const v8h*)base;                           // K = off..off+7
            v8h hi = *(const v8h*)(base + 8);                     // K = off+8..+15
            v16h af;
            #pragma unroll
            for (int i = 0; i < 8; ++i) { af[i] = lo[i]; af[i + 8] = hi[i]; }
            acc[rb] = __builtin_amdgcn_wmma_f32_16x16x32_f16(
                false, af, false, bf, (short)0, acc[rb], false, false);
        }

        // commit next chunk to the other LDS buffer, then one barrier/iter
        if (it + 1 < NCH) ((v8h*)vtile[nxt])[cstage * 4 + chs] = vreg;
        __syncthreads();
    }

    // complete column sums S_j (halves live in lanes l and l^16)
    const float ssum = (ssa[0] + ssa[1]) + (ssa[2] + ssa[3]);
    const float S  = ssum + __shfl_xor(ssum, 16, 32);
    const float fj = 1.0f / (1.5f * S);

    // write T transposed (f16) to per-wave LDS region: TtS[wave][j][c]
    #pragma unroll
    for (int rb = 0; rb < 4; ++rb) {
        v8h t8;
        #pragma unroll
        for (int r = 0; r < 8; ++r) t8[r] = (_Float16)acc[rb][r];
        *(v8h*)(TtS + wave * 1024 + l15 * 64 + rb * 16 + lhi * 8) = t8;
    }
    __syncthreads();

    // fused gamma GEMM: final = Wg(64x64) @ T(64x16), two K-steps of 32
    v8f acc2[4] = {};
    #pragma unroll
    for (int ks = 0; ks < 2; ++ks) {
        const int kb = ks * 32;
        const _Float16* tb = TtS + wave * 1024 + l15 * 64 + kb + lhi * 16;
        v8h blo = *(const v8h*)tb;
        v8h bhi = *(const v8h*)(tb + 8);
        v16h bf2;
        #pragma unroll
        for (int i = 0; i < 8; ++i) { bf2[i] = blo[i]; bf2[i + 8] = bhi[i]; }
        #pragma unroll
        for (int rb = 0; rb < 4; ++rb) {
            const int o = rb * 16 + l15;
            const _Float16* ab = wgS + o * CCH + kb + lhi * 16;
            v8h lo = *(const v8h*)ab;
            v8h hi = *(const v8h*)(ab + 8);
            v16h af;
            #pragma unroll
            for (int i = 0; i < 8; ++i) { af[i] = lo[i]; af[i + 8] = hi[i]; }
            acc2[rb] = __builtin_amdgcn_wmma_f32_16x16x32_f16(
                false, af, false, bf2, (short)0, acc2[rb], false, false);
        }
    }

    // epilogue: scale by f_j, add bg, store (B,C,N) f32
    #pragma unroll
    for (int rb = 0; rb < 4; ++rb) {
        #pragma unroll
        for (int r = 0; r < 8; ++r) {
            const int o = rb * 16 + lhi * 8 + r;
            out[(((size_t)(b * CCH + o)) << 12) + j] = fmaf(acc2[rb][r], fj, bgS[o]);
        }
    }
}

// ---------------------------------------------------------------------------
extern "C" void kernel_launch(void* const* d_in, const int* in_sizes, int n_in,
                              void* d_out, int out_size, void* d_ws, size_t ws_size,
                              hipStream_t stream) {
    const float* x   = (const float*)d_in[0];
    const float* Wq  = (const float*)d_in[1];
    const float* bq  = (const float*)d_in[2];
    const float* Wk  = (const float*)d_in[3];
    const float* bk  = (const float*)d_in[4];
    const float* wcq = (const float*)d_in[5];
    const float* wck = (const float*)d_in[6];
    const float* Wv  = (const float*)d_in[7];
    const float* bv  = (const float*)d_in[8];
    const float* Wg  = (const float*)d_in[9];
    const float* bg  = (const float*)d_in[10];
    float* out = (float*)d_out;

    char* ws = (char*)d_ws;
    float*    uq   = (float*)(ws + 0);
    float*    uk   = (float*)(ws + 256);
    float*    cqk  = (float*)(ws + 512);
    _Float16* wg16 = (_Float16*)(ws + 1024);                  // 8 KB
    float2*   aeB  = (float2*)(ws + 16384);                   // B*N*8 = 128 KB
    float2*   deB  = (float2*)(ws + 16384 + 131072);          // 128 KB
    _Float16* vB   = (_Float16*)(ws + 16384 + 262144);        // B*C*N f16 = 2 MB

    attn_setup<<<1, 64, 0, stream>>>(Wq, bq, Wk, bk, wcq, wck, Wg,
                                     uq, uk, cqk, wg16);

    attn_proj<<<dim3(NPIX / 256, NB), 256, 0, stream>>>(
        x, Wv, bv, uq, uk, cqk, aeB, deB, vB);

    attn_main<<<dim3(NPIX / 128, NB), 256, 0, stream>>>(
        aeB, deB, vB, wg16, bg, out);
}
